// DenseMatchingHead_29557964931183
// MI455X (gfx1250) — matrix operand order repaired and checked by reading, not compiled, not run on hardware
//
#include <hip/hip_runtime.h>
#include <hip/hip_bf16.h>

typedef __attribute__((ext_vector_type(16))) _Float16 v16h;
typedef __attribute__((ext_vector_type(8)))  _Float16 v8h;
typedef __attribute__((ext_vector_type(4)))  _Float16 v4h;
typedef __attribute__((ext_vector_type(8)))  float    v8f;

#define BB    4
#define NN    4096
#define CC    128
#define KSEL  100
#define INV_T 10.0f

#define BTI   128            // block tile rows (i)
#define BTJ   64             // block tile cols (j)
#define LDSP  (CC + 8)       // padded LDS row pitch in halfs (272B: bank-conflict-free)

// ---------------- fragment loaders (ISA 7.12.2 layouts) ----------------
// A 16(M)x32(K) f16: lane m = lane&15; kb=(lane>>4)*8;
//   elems 0..7 -> K = kb+0..7 ; elems 8..15 -> K = 16+kb+0..7
__device__ __forceinline__ v16h ldA(const _Float16* rowp, int kb) {
  union { v16h v; v8h h[2]; } u;
  u.h[0] = *(const v8h*)(rowp + kb);
  u.h[1] = *(const v8h*)(rowp + 16 + kb);
  return u.v;
}
// B 32(K)x16(N) f16 with B[k][n] = f2n[j0+n][k0+k]: lane n = lane&15;
//   Kbase=(lane>>4)*8; elems 0..7 -> K = Kbase+0..7 ; elems 8..15 -> K = 16+Kbase+0..7
__device__ __forceinline__ v16h ldB(const _Float16* rowp, int Kbase) {
  union { v16h v; v8h h[2]; } u;
  u.h[0] = *(const v8h*)(rowp + Kbase);
  u.h[1] = *(const v8h*)(rowp + 16 + Kbase);
  return u.v;
}

// ---------------- cooperative LDS staging of the block tile ----------------
// A: 128 rows x 128 halfs, B: 64 rows x 128 halfs; rows padded to LDSP.
__device__ __forceinline__ void stage_tiles(_Float16* sA, _Float16* sB,
                                            const _Float16* A, const _Float16* Bm,
                                            int gi0, int gj0) {
  const int t = threadIdx.x;
#pragma unroll
  for (int c = 0; c < 8; ++c) {            // A: 2048 x 16B chunks
    const int chunk = t + c * 256;
    const int row = chunk >> 4, col = (chunk & 15) * 8;
    *(v8h*)(sA + row * LDSP + col) =
        *(const v8h*)(A + (size_t)(gi0 + row) * CC + col);
  }
#pragma unroll
  for (int c = 0; c < 4; ++c) {            // B: 1024 x 16B chunks
    const int chunk = t + c * 256;
    const int row = chunk >> 4, col = (chunk & 15) * 8;
    *(v8h*)(sB + row * LDSP + col) =
        *(const v8h*)(Bm + (size_t)(gj0 + row) * CC + col);
  }
  __syncthreads();
}

// Each wave computes a 32x32 corr sub-tile out of LDS-staged operands.
__device__ __forceinline__ void corr_tile_lds(const _Float16* sA, const _Float16* sB,
                                              int li0, int lj0, v8f acc[2][2]) {
  const int lane = threadIdx.x & 31;
  const int ln   = lane & 15;
  const int kbA  = (lane >> 4) * 8;    // A lane-half K interleave
  const int kbB  = (lane >> 4) * 8;    // B lane-half K interleave
  const _Float16* pa = sA + (li0 + ln) * LDSP;
  const _Float16* pb = sB + (lj0 + ln) * LDSP;
#pragma unroll
  for (int k0 = 0; k0 < CC; k0 += 32) {
    v16h a0 = ldA(pa + k0,             kbA);
    v16h a1 = ldA(pa + 16 * LDSP + k0, kbA);
    v16h b0 = ldB(pb + k0,             kbB);
    v16h b1 = ldB(pb + 16 * LDSP + k0, kbB);
    acc[0][0] = __builtin_amdgcn_wmma_f32_16x16x32_f16(false, a0, false, b0, (short)0, acc[0][0], false, false);
    acc[0][1] = __builtin_amdgcn_wmma_f32_16x16x32_f16(false, a0, false, b1, (short)0, acc[0][1], false, false);
    acc[1][0] = __builtin_amdgcn_wmma_f32_16x16x32_f16(false, a1, false, b0, (short)0, acc[1][0], false, false);
    acc[1][1] = __builtin_amdgcn_wmma_f32_16x16x32_f16(false, a1, false, b1, (short)0, acc[1][1], false, false);
  }
}

// ---------------- kernel 1: cosine-normalize + f32->f16 ----------------
__global__ void __launch_bounds__(256)
normalize_f16_kernel(const float* __restrict__ f1, const float* __restrict__ f2,
                     _Float16* __restrict__ o1, _Float16* __restrict__ o2) {
  const int lane = threadIdx.x & 31;
  const int wave = threadIdx.x >> 5;
  const int d    = blockIdx.x * 8 + wave;          // 0 .. BB*NN-1
  const float* in  = blockIdx.y ? f2 : f1;
  _Float16*    out = blockIdx.y ? o2 : o1;
  float4 v = ((const float4*)(in + (size_t)d * CC))[lane];
  float s = v.x * v.x + v.y * v.y + v.z * v.z + v.w * v.w;
  s += __shfl_xor(s, 1);  s += __shfl_xor(s, 2);  s += __shfl_xor(s, 4);
  s += __shfl_xor(s, 8);  s += __shfl_xor(s, 16);
  const float scale = 1.0f / (sqrtf(s) + 1e-8f);
  v4h h;
  h[0] = (_Float16)(v.x * scale);
  h[1] = (_Float16)(v.y * scale);
  h[2] = (_Float16)(v.z * scale);
  h[3] = (_Float16)(v.w * scale);
  *(v4h*)(out + (size_t)d * CC + lane * 4) = h;
}

// ---------------- kernel 2: init stats ----------------
__global__ void init_stats_kernel(float* row_sum, float* col_sum,
                                  unsigned long long* rowbest) {
  const int i = blockIdx.x * blockDim.x + threadIdx.x;
  if (i < BB * NN) { row_sum[i] = 0.f; col_sum[i] = 0.f; rowbest[i] = 0ull; }
}

// ---------------- kernel 3: GEMM pass 1 -> row/col exp-sums ----------------
// corr/T in [-10,10] (cosine / 0.1) so exp() is overflow-safe w/o max subtraction.
__global__ void __launch_bounds__(256)
corr_stats_kernel(const _Float16* __restrict__ f1h, const _Float16* __restrict__ f2h,
                  float* __restrict__ row_sum, float* __restrict__ col_sum) {
  __shared__ _Float16 sA[BTI * LDSP];   // 34816 B
  __shared__ _Float16 sB[BTJ * LDSP];   // 17408 B
  const int wave = threadIdx.x >> 5, lane = threadIdx.x & 31;
  const int b   = blockIdx.z;
  const int gi0 = blockIdx.y * BTI;
  const int gj0 = blockIdx.x * BTJ;
  const _Float16* A  = f1h + (size_t)b * NN * CC;
  const _Float16* Bm = f2h + (size_t)b * NN * CC;
  stage_tiles(sA, sB, A, Bm, gi0, gj0);

  const int li0 = (wave >> 1) * 32;     // wave sub-tile origin (block-local)
  const int lj0 = (wave & 1)  * 32;
  v8f acc[2][2] = {};
  corr_tile_lds(sA, sB, li0, lj0, acc);

  const int half = lane >> 4;
  const int ln   = lane & 15;
  float ev[2][2][8];
#pragma unroll
  for (int si = 0; si < 2; ++si)
#pragma unroll
    for (int sj = 0; sj < 2; ++sj)
#pragma unroll
      for (int t = 0; t < 8; ++t)
        ev[si][sj][t] = __expf(acc[si][sj][t] * INV_T);

  // row sums: value at (vgpr t, lane) -> row gi0+li0+si*16+half*8+t, col gj0+lj0+sj*16+ln
#pragma unroll
  for (int si = 0; si < 2; ++si)
#pragma unroll
    for (int t = 0; t < 8; ++t) {
      float s = ev[si][0][t] + ev[si][1][t];
      s += __shfl_xor(s, 1); s += __shfl_xor(s, 2);
      s += __shfl_xor(s, 4); s += __shfl_xor(s, 8);
      if (ln == 0)
        atomicAdd(&row_sum[b * NN + gi0 + li0 + si * 16 + half * 8 + t], s);
    }
  // col sums
#pragma unroll
  for (int sj = 0; sj < 2; ++sj) {
    float c = 0.f;
#pragma unroll
    for (int si = 0; si < 2; ++si)
#pragma unroll
      for (int t = 0; t < 8; ++t) c += ev[si][sj][t];
    c += __shfl_xor(c, 16);
    if (lane < 16)
      atomicAdd(&col_sum[b * NN + gj0 + lj0 + sj * 16 + ln], c);
  }
}

// ---------------- kernel 4: GEMM pass 2 -> write P + row best ----------------
__global__ void __launch_bounds__(256)
corr_write_kernel(const _Float16* __restrict__ f1h, const _Float16* __restrict__ f2h,
                  const float* __restrict__ row_sum, const float* __restrict__ col_sum,
                  unsigned long long* __restrict__ rowbest, float* __restrict__ P) {
  __shared__ _Float16 sA[BTI * LDSP];
  __shared__ _Float16 sB[BTJ * LDSP];
  const int wave = threadIdx.x >> 5, lane = threadIdx.x & 31;
  const int b   = blockIdx.z;
  const int gi0 = blockIdx.y * BTI;
  const int gj0 = blockIdx.x * BTJ;
  const _Float16* A  = f1h + (size_t)b * NN * CC;
  const _Float16* Bm = f2h + (size_t)b * NN * CC;
  stage_tiles(sA, sB, A, Bm, gi0, gj0);

  const int li0 = (wave >> 1) * 32;
  const int lj0 = (wave & 1)  * 32;
  v8f acc[2][2] = {};
  corr_tile_lds(sA, sB, li0, lj0, acc);

  const int half = lane >> 4;
  const int ln   = lane & 15;
  const int j0   = gj0 + lj0;
  const float cs0 = col_sum[b * NN + j0 + ln];
  const float cs1 = col_sum[b * NN + j0 + 16 + ln];
  const unsigned jA = (unsigned)(j0 + ln);
  const unsigned jB = (unsigned)(j0 + 16 + ln);
  const size_t pbase = (size_t)b * NN * NN;

#pragma unroll
  for (int si = 0; si < 2; ++si)
#pragma unroll
    for (int t = 0; t < 8; ++t) {
      const int i  = gi0 + li0 + si * 16 + half * 8 + t;
      const float rs = row_sum[b * NN + i];
      const float e0 = __expf(acc[si][0][t] * INV_T);
      const float e1 = __expf(acc[si][1][t] * INV_T);
      // P = e / sqrt(rowsum * colsum)
      const float p0 = e0 * rsqrtf(rs * cs0);
      const float p1 = e1 * rsqrtf(rs * cs1);
      float* rp = P + pbase + (size_t)i * NN;
      __builtin_nontemporal_store(p0, rp + jA);
      __builtin_nontemporal_store(p1, rp + jB);
      // packed (value, ~j): max picks larger P, smaller j on ties (jnp.argmax)
      float mv; unsigned mj;
      if (p1 > p0) { mv = p1; mj = jB; } else { mv = p0; mj = jA; }
      unsigned long long pk =
          ((unsigned long long)__float_as_uint(mv) << 32) |
          (unsigned long long)(0xFFFFFFFFu - mj);
#pragma unroll
      for (int m = 1; m < 16; m <<= 1) {
        unsigned long long o = __shfl_xor(pk, m);
        if (o > pk) pk = o;
      }
      if (ln == 0)
        atomicMax(&rowbest[b * NN + i], pk);
    }
}

// ---------------- kernel 5: top-K rows by max confidence ----------------
// rank(i) = #{k: mv[k] > mv[i]} + #{k<i: mv[k]==mv[i]}  (stable argsort order)
__global__ void __launch_bounds__(256)
topk_kernel(const unsigned long long* __restrict__ rowbest,
            float* __restrict__ out_matches) {
  __shared__ float mv[NN];
  const int b = blockIdx.y;
  for (int i = threadIdx.x; i < NN; i += 256)
    mv[i] = __uint_as_float((unsigned)(rowbest[b * NN + i] >> 32));
  __syncthreads();
  const int i = blockIdx.x * 256 + threadIdx.x;
  const float v = mv[i];
  int rank = 0;
  for (int k = 0; k < NN; ++k) {
    const float u = mv[k];
    rank += (u > v) || (u == v && k < i);
  }
  if (rank < KSEL) {
    const unsigned long long pk = rowbest[b * NN + i];
    const int j = (int)(0xFFFFFFFFu - (unsigned)(pk & 0xFFFFFFFFu));
    float* m = out_matches + ((size_t)b * KSEL + rank) * 2;
    m[0] = (float)i;
    m[1] = (float)j;
  }
}

// ---------------- launch ----------------
extern "C" void kernel_launch(void* const* d_in, const int* in_sizes, int n_in,
                              void* d_out, int out_size, void* d_ws, size_t ws_size,
                              hipStream_t stream) {
  const float* feat1 = (const float*)d_in[0];
  const float* feat2 = (const float*)d_in[1];
  float* out = (float*)d_out;

  char* ws = (char*)d_ws;
  _Float16* f1h = (_Float16*)(ws);                             // 4 MB
  _Float16* f2h = (_Float16*)(ws + (size_t)4 * 1024 * 1024);   // 4 MB
  float* row_sum = (float*)(ws + (size_t)8 * 1024 * 1024);                  // 64 KB
  float* col_sum = (float*)(ws + (size_t)8 * 1024 * 1024 + 64 * 1024);      // 64 KB
  unsigned long long* rowbest =
      (unsigned long long*)(ws + (size_t)8 * 1024 * 1024 + 128 * 1024);     // 128 KB

  // 1) normalize + convert both feature maps to f16 (grid.y selects tensor)
  normalize_f16_kernel<<<dim3(BB * NN / 8, 2), 256, 0, stream>>>(feat1, feat2, f1h, f2h);

  // 2) zero softmax accumulators / row-best
  init_stats_kernel<<<(BB * NN + 255) / 256, 256, 0, stream>>>(row_sum, col_sum, rowbest);

  // 3) WMMA GEMM pass 1 (LDS-staged): exp-sum stats (rows + cols)
  dim3 ggrid(NN / BTJ, NN / BTI, BB);
  corr_stats_kernel<<<ggrid, 256, 0, stream>>>(f1h, f2h, row_sum, col_sum);

  // 4) WMMA GEMM pass 2 (LDS-staged): write P (nontemporal) + fused row argmax
  corr_write_kernel<<<ggrid, 256, 0, stream>>>(f1h, f2h, row_sum, col_sum, rowbest, out);

  // 5) top-100 match list appended after P
  float* out_matches = out + (size_t)BB * NN * NN;
  topk_kernel<<<dim3(NN / 256, BB), 256, 0, stream>>>(rowbest, out_matches);
}
